// GFRN_encoder_10256381903666
// MI455X (gfx1250) — compile-verified, matrix operand order checked by reading, not run on hardware
//
#include <hip/hip_runtime.h>
#include <math.h>

#define Bsz 8
#define ROI 90
#define Tt  200
#define Hh  3
#define Ff  64
#define DHh 64
#define Rr  270
#define G3  192
#define BNE 1e-5f
#define MP  96
#define KP  96

typedef __attribute__((ext_vector_type(16))) _Float16 v16h;
typedef __attribute__((ext_vector_type(4)))  _Float16 v4h;
typedef __attribute__((ext_vector_type(2)))  _Float16 v2h;
typedef __attribute__((ext_vector_type(8)))  float    v8f;
typedef __attribute__((ext_vector_type(4)))  float    v4f;
typedef __attribute__((ext_vector_type(2)))  float    v2f;
typedef __attribute__((ext_vector_type(4)))  int      v4i;

#if defined(__has_builtin)
#  if __has_builtin(__builtin_amdgcn_global_load_async_to_lds_b128) && \
      __has_builtin(__builtin_amdgcn_s_wait_asynccnt)
#    define HAVE_ASYNC_LDS 1
#  endif
#endif
#ifndef HAVE_ASYNC_LDS
#  define HAVE_ASYNC_LDS 0
#endif

#if HAVE_ASYNC_LDS
typedef __attribute__((address_space(1))) v4i* gv4i_p;   // global (AS1) 16B vector ptr
typedef __attribute__((address_space(3))) v4i* lv4i_p;   // LDS    (AS3) 16B vector ptr
__device__ __forceinline__ void async_cp16(const float* g, float* l) {
  __builtin_amdgcn_global_load_async_to_lds_b128((gv4i_p)g, (lv4i_p)l, 0, 0);
}
#endif

__device__ __forceinline__ v8f wmma_f16(v16h a, v16h b, v8f c) {
  return __builtin_amdgcn_wmma_f32_16x16x32_f16(false, a, false, b, (short)0, c, false, false);
}

// Fragment loader for row-major LDS tiles. For A: p=[M][K], row0=m0.
// For B stored TRANSPOSED (Bt[n][k]): p=[N][K], row0=n0.  Per lane: 2x contiguous
// 8-half runs -> ds_load_b128 pairs.
__device__ __forceinline__ v16h frag_ld(const _Float16* p, int ldk, int row0, int k0, int lane) {
  const _Float16* row = p + (size_t)(row0 + (lane & 15)) * ldk + k0 + ((lane >> 4) << 3);
  v16h v;
#pragma unroll
  for (int e = 0; e < 8; ++e) v[e] = row[e];
#pragma unroll
  for (int e = 0; e < 8; ++e) v[8 + e] = row[16 + e];
  return v;
}

__global__ void k_zero(float* p, int n) {
  int i = blockIdx.x * 256 + threadIdx.x;
  if (i < n) p[i] = 0.f;
}

// ---- K1: agg = mask@x ; v = agg + eps*x ; z1 = v@W1 + b1 ; BN1 partial sums ----
__global__ __launch_bounds__(256)
void k1_gnn1(const float* __restrict__ x, const float* __restrict__ a,
             const float* __restrict__ eps_param,
             const float* __restrict__ W1, const float* __restrict__ b1,
             float* __restrict__ z1, float* __restrict__ gsum, float* __restrict__ gsq) {
  __shared__ _Float16 Msk[MP][KP];   // A1  [m][k=j]
  __shared__ _Float16 XsT[Ff][KP];   // B1^T [n=f][k=j]
  __shared__ _Float16 Vs[MP][Ff];    // A2  [m][k=f]
  __shared__ _Float16 WsT[Ff][Ff];   // B2^T [n=g][k=f]
  __shared__ float redS[Ff], redQ[Ff];

  int bid = blockIdx.x;                 // (b*T + t)*H + h
  int h = bid % Hh; int bt = bid / Hh;
  int t = bt % Tt;  int b = bt / Tt;
  int tid = threadIdx.x, lane = tid & 31, wave = tid >> 5;

  if (tid < Ff) { redS[tid] = 0.f; redQ[tid] = 0.f; }

  for (int i = tid; i < KP * (Ff / 4); i += 256) {
    int j = i >> 4, f4 = (i & 15) << 2;
    v4f xv = {0.f, 0.f, 0.f, 0.f};
    if (j < ROI) xv = *(const v4f*)&x[((((size_t)b * ROI + j) * Tt + t) * Hh + h) * Ff + f4];
#pragma unroll
    for (int e = 0; e < 4; ++e) XsT[f4 + e][j] = (_Float16)xv[e];
  }
  for (int i = tid; i < MP * (KP / 2); i += 256) {
    int rr = i / (KP / 2), c2 = (i % (KP / 2)) * 2;
    v2h mm = {(_Float16)0.f, (_Float16)0.f};
    if (rr < ROI && c2 < ROI) {
      v2f av = *(const v2f*)&a[((((size_t)b * ROI + rr) * Tt + t) * Hh + h) * ROI + c2];
      mm[0] = (av[0] != 0.f) ? (_Float16)1.f : (_Float16)0.f;
      mm[1] = (av[1] != 0.f) ? (_Float16)1.f : (_Float16)0.f;
    }
    *(v2h*)&Msk[rr][c2] = mm;
  }
  for (int i = tid; i < Ff * (Ff / 4); i += 256) {
    int f = i >> 4, g4 = (i & 15) << 2;
    v4f wv = *(const v4f*)&W1[((size_t)h * Ff + f) * Ff + g4];
#pragma unroll
    for (int e = 0; e < 4; ++e) WsT[g4 + e][f] = (_Float16)wv[e];
  }
  __syncthreads();

  float epsh = eps_param[h];
  for (int s = 0; s < 3; ++s) {
    int tile = wave * 3 + s;
    int mt = tile >> 2, nt = tile & 3;
    v8f c = {};
#pragma unroll
    for (int kt = 0; kt < 3; ++kt) {
      v16h af = frag_ld(&Msk[0][0], KP, mt * 16, kt * 32, lane);
      v16h bf = frag_ld(&XsT[0][0], KP, nt * 16, kt * 32, lane);
      c = wmma_f16(af, bf, c);
    }
    int n = nt * 16 + (lane & 15);
#pragma unroll
    for (int j = 0; j < 8; ++j) {
      int m = mt * 16 + j + ((lane >> 4) << 3);
      float v = c[j] + epsh * (float)XsT[n][m];
      Vs[m][n] = (_Float16)v;
    }
  }
  __syncthreads();
  for (int s = 0; s < 3; ++s) {
    int tile = wave * 3 + s;
    int mt = tile >> 2, nt = tile & 3;
    v8f c = {};
#pragma unroll
    for (int kt = 0; kt < 2; ++kt) {
      v16h af = frag_ld(&Vs[0][0],  Ff, mt * 16, kt * 32, lane);
      v16h bf = frag_ld(&WsT[0][0], Ff, nt * 16, kt * 32, lane);
      c = wmma_f16(af, bf, c);
    }
    int n = nt * 16 + (lane & 15);
    float bn = b1[h * Ff + n];
    float ls = 0.f, lq = 0.f;
#pragma unroll
    for (int j = 0; j < 8; ++j) {
      int m = mt * 16 + j + ((lane >> 4) << 3);
      if (m < ROI) {
        float z = c[j] + bn;
        z1[((((size_t)b * ROI + m) * Tt + t) * Hh + h) * Ff + n] = z;
        ls += z; lq += z * z;
      }
    }
    atomicAdd(&redS[n], ls);
    atomicAdd(&redQ[n], lq);
  }
  __syncthreads();
  if (tid < Ff) {
    atomicAdd(&gsum[h * Ff + tid], redS[tid]);
    atomicAdd(&gsq[h * Ff + tid],  redQ[tid]);
  }
}

// ---- BN finalize: affine A = g*rstd, Bc = be - A*mean ----
__global__ void k_bnfin(const float* gsum, const float* gsq,
                        const float* g, const float* be,
                        float* A, float* Bc, float invN) {
  int i = threadIdx.x;                  // 0..191
  float m = gsum[i] * invN;
  float v = gsq[i] * invN - m * m;
  float rs = rsqrtf(v + BNE);
  float sc = g[i] * rs;
  A[i]  = sc;
  Bc[i] = be[i] - sc * m;
}

// ---- K3: v = elu(A1*z1+B1) ; z2 = v@W2 + b2 ; BN2 partial sums ----
__global__ __launch_bounds__(256)
void k3_gnn2(const float* __restrict__ z1,
             const float* __restrict__ A1, const float* __restrict__ B1,
             const float* __restrict__ W2, const float* __restrict__ b2,
             float* __restrict__ z2, float* __restrict__ gsum, float* __restrict__ gsq) {
  __shared__ _Float16 Vs[MP][Ff];      // A
  __shared__ _Float16 WsT[Ff][Ff];     // B^T [g][f]
  __shared__ float redS[Ff], redQ[Ff];
  int bid = blockIdx.x;
  int h = bid % Hh; int bt = bid / Hh;
  int t = bt % Tt;  int b = bt / Tt;
  int tid = threadIdx.x, lane = tid & 31, wave = tid >> 5;
  if (tid < Ff) { redS[tid] = 0.f; redQ[tid] = 0.f; }

  for (int i = tid; i < MP * (Ff / 4); i += 256) {
    int m = i >> 4, f4 = (i & 15) << 2;
    v4h o = {(_Float16)0.f, (_Float16)0.f, (_Float16)0.f, (_Float16)0.f};
    if (m < ROI) {
      v4f zv = *(const v4f*)&z1[((((size_t)b * ROI + m) * Tt + t) * Hh + h) * Ff + f4];
#pragma unroll
      for (int e = 0; e < 4; ++e) {
        float q = A1[h * Ff + f4 + e] * zv[e] + B1[h * Ff + f4 + e];
        o[e] = (_Float16)(q > 0.f ? q : (__expf(q) - 1.f));   // ELU
      }
    }
    *(v4h*)&Vs[m][f4] = o;
  }
  for (int i = tid; i < Ff * (Ff / 4); i += 256) {
    int f = i >> 4, g4 = (i & 15) << 2;
    v4f wv = *(const v4f*)&W2[((size_t)h * Ff + f) * Ff + g4];
#pragma unroll
    for (int e = 0; e < 4; ++e) WsT[g4 + e][f] = (_Float16)wv[e];
  }
  __syncthreads();

  for (int s = 0; s < 3; ++s) {
    int tile = wave * 3 + s;
    int mt = tile >> 2, nt = tile & 3;
    v8f c = {};
#pragma unroll
    for (int kt = 0; kt < 2; ++kt) {
      v16h af = frag_ld(&Vs[0][0],  Ff, mt * 16, kt * 32, lane);
      v16h bf = frag_ld(&WsT[0][0], Ff, nt * 16, kt * 32, lane);
      c = wmma_f16(af, bf, c);
    }
    int n = nt * 16 + (lane & 15);
    float bn = b2[h * Ff + n];
    float ls = 0.f, lq = 0.f;
#pragma unroll
    for (int j = 0; j < 8; ++j) {
      int m = mt * 16 + j + ((lane >> 4) << 3);
      if (m < ROI) {
        float z = c[j] + bn;
        z2[((((size_t)b * ROI + m) * Tt + t) * Hh + h) * Ff + n] = z;
        ls += z; lq += z * z;
      }
    }
    atomicAdd(&redS[n], ls);
    atomicAdd(&redQ[n], lq);
  }
  __syncthreads();
  if (tid < Ff) {
    atomicAdd(&gsum[h * Ff + tid], redS[tid]);
    atomicAdd(&gsq[h * Ff + tid],  redQ[tid]);
  }
}

// ---- K5a: GI[r][t][b][g] = relu(bn2(z2)) @ Wih^T + bih ----
__global__ __launch_bounds__(256)
void k5a_gi(const float* __restrict__ z2,
            const float* __restrict__ A2, const float* __restrict__ B2,
            const float* __restrict__ Wih, const float* __restrict__ bih,
            float* __restrict__ GI) {
  __shared__ _Float16 Ys[64][Ff];       // A: 8 t x 8 b rows
  __shared__ _Float16 Wt[G3][Ff];       // B^T [go][f]  == Wih[r] rows, straight copy
  int r = blockIdx.x, tc = blockIdx.y;  // r 0..269, tc 0..24
  int roi = r / Hh, hh = r % Hh;
  int tid = threadIdx.x, lane = tid & 31, wave = tid >> 5;

  for (int i = tid; i < 64 * (Ff / 4); i += 256) {
    int row = i >> 4, f4 = (i & 15) << 2;
    int tl = row >> 3, b = row & 7;
    int t = tc * 8 + tl;
    v4f zv = *(const v4f*)&z2[((((size_t)b * ROI + roi) * Tt + t) * Hh + hh) * Ff + f4];
    v4h o;
#pragma unroll
    for (int e = 0; e < 4; ++e) {
      float q = A2[hh * Ff + f4 + e] * zv[e] + B2[hh * Ff + f4 + e];
      o[e] = (_Float16)(q > 0.f ? q : 0.f);   // relu(elu(q)) == max(q,0)
    }
    *(v4h*)&Ys[row][f4] = o;
  }
  for (int i = tid; i < G3 * (Ff / 4); i += 256) {
    int go = i >> 4, f4 = (i & 15) << 2;
    v4f wv = *(const v4f*)&Wih[((size_t)r * G3 + go) * Ff + f4];
    v4h o = {(_Float16)wv[0], (_Float16)wv[1], (_Float16)wv[2], (_Float16)wv[3]};
    *(v4h*)&Wt[go][f4] = o;
  }
  __syncthreads();

  for (int s = 0; s < 6; ++s) {                   // 4x12 tiles, 8 waves x 6
    int tile = wave * 6 + s;
    int mt = tile / 12, nt = tile % 12;
    v8f c = {};
#pragma unroll
    for (int kt = 0; kt < 2; ++kt) {
      v16h af = frag_ld(&Ys[0][0], Ff, mt * 16, kt * 32, lane);
      v16h bf = frag_ld(&Wt[0][0], Ff, nt * 16, kt * 32, lane);
      c = wmma_f16(af, bf, c);
    }
    int n = nt * 16 + (lane & 15);
    float bn = bih[(size_t)r * G3 + n];
#pragma unroll
    for (int j = 0; j < 8; ++j) {
      int m = mt * 16 + j + ((lane >> 4) << 3);
      int tl = m >> 3, b = m & 7;
      int t = tc * 8 + tl;
      GI[(((size_t)r * Tt + t) * Bsz + b) * G3 + n] = c[j] + bn;
    }
  }
}

// ---- K5b: sequential GRU recurrence; hoisted weight fragments + async-LDS
//      double-buffered gate stream ----
__global__ __launch_bounds__(128)
void k5b_gru(const float* __restrict__ GI,
             const float* __restrict__ Whh, const float* __restrict__ bhh,
             float* __restrict__ out) {
  __shared__ _Float16 Wt[G3][DHh];      // B^T [go][d] == Whh[r] rows
  __shared__ _Float16 Ha[16][DHh];      // A: h state, rows 8..15 stay zero
  __shared__ float Gh[G3][Bsz];         // gh, column-major for vectorized stores
  __shared__ float Hf[Bsz][DHh];
#if HAVE_ASYNC_LDS
  __shared__ float GIb[2][Bsz * G3];    // double-buffered gate slabs
#endif
  int r = blockIdx.x;
  int roi = r / Hh, hh = r % Hh;
  int tid = threadIdx.x, lane = tid & 31, wave = tid >> 5;

  for (int i = tid; i < G3 * (DHh / 4); i += 128) {
    int go = i >> 4, d4 = (i & 15) << 2;
    v4f wv = *(const v4f*)&Whh[((size_t)r * G3 + go) * DHh + d4];
    v4h o = {(_Float16)wv[0], (_Float16)wv[1], (_Float16)wv[2], (_Float16)wv[3]};
    *(v4h*)&Wt[go][d4] = o;
  }
  for (int i = tid; i < 16 * DHh / 4; i += 128) {
    v4h z = {(_Float16)0.f, (_Float16)0.f, (_Float16)0.f, (_Float16)0.f};
    *(v4h*)&(&Ha[0][0])[i * 4] = z;
  }
  for (int i = tid; i < Bsz * DHh; i += 128) (&Hf[0][0])[i] = 0.f;
  __syncthreads();

  // hoist loop-invariant B fragments + bias into registers (held for all 200 steps)
  v16h bfr[3][2];
  float bias[3];
  int ncol[3];
#pragma unroll
  for (int s = 0; s < 3; ++s) {
    int nt = wave * 3 + s;
    ncol[s] = nt * 16 + (lane & 15);
#pragma unroll
    for (int kt = 0; kt < 2; ++kt)
      bfr[s][kt] = frag_ld(&Wt[0][0], DHh, nt * 16, kt * 32, lane);
    bias[s] = bhh[(size_t)r * G3 + ncol[s]];
  }
  __syncthreads();

#if HAVE_ASYNC_LDS
  {  // prefetch gate slab t=0 into buffer 0
    const float* g0 = GI + (size_t)r * Tt * Bsz * G3;
#pragma unroll
    for (int q = 0; q < 3; ++q) {
      int off = (tid + q * 128) * 4;
      async_cp16(g0 + off, &GIb[0][off]);
    }
  }
#endif

  int eb = tid >> 4;                  // elementwise batch index (0..7)
  int ed4 = (tid & 15) << 2;          // elementwise d-quad base

  for (int t = 0; t < Tt; ++t) {
    // h(f32) -> A fragment (f16)
    {
      v4f hv = *(const v4f*)&Hf[eb][ed4];
      v4h o = {(_Float16)hv[0], (_Float16)hv[1], (_Float16)hv[2], (_Float16)hv[3]};
      *(v4h*)&Ha[eb][ed4] = o;
    }
#if HAVE_ASYNC_LDS
    __builtin_amdgcn_s_wait_asynccnt(0);   // gate slab for step t landed in LDS
#endif
    __syncthreads();
#if HAVE_ASYNC_LDS
    if (t + 1 < Tt) {                      // start streaming slab t+1 into other buffer
      const float* gn = GI + ((size_t)r * Tt + t + 1) * Bsz * G3;
#pragma unroll
      for (int q = 0; q < 3; ++q) {
        int off = (tid + q * 128) * 4;
        async_cp16(gn + off, &GIb[(t + 1) & 1][off]);
      }
    }
#endif
    v16h af0 = frag_ld(&Ha[0][0], DHh, 0, 0,  lane);
    v16h af1 = frag_ld(&Ha[0][0], DHh, 0, 32, lane);
#pragma unroll
    for (int s = 0; s < 3; ++s) {
      v8f c = {};
      c = wmma_f16(af0, bfr[s][0], c);
      c = wmma_f16(af1, bfr[s][1], c);
      if (lane < 16) {                   // lanes 0-15 hold rows m=0..7 (valid batch)
        float bv = bias[s];
        v4f lo = {c[0] + bv, c[1] + bv, c[2] + bv, c[3] + bv};
        v4f hi = {c[4] + bv, c[5] + bv, c[6] + bv, c[7] + bv};
        *(v4f*)&Gh[ncol[s]][0] = lo;
        *(v4f*)&Gh[ncol[s]][4] = hi;
      }
    }
    __syncthreads();
#if HAVE_ASYNC_LDS
    const float* gb = &GIb[t & 1][eb * G3];
#else
    const float* gb = GI + (((size_t)r * Tt + t) * Bsz + eb) * G3;
#endif
    {
      v4f ir4 = *(const v4f*)&gb[ed4];
      v4f iz4 = *(const v4f*)&gb[64 + ed4];
      v4f in4 = *(const v4f*)&gb[128 + ed4];
      v4f hp4 = *(const v4f*)&Hf[eb][ed4];
      v4f hv4;
#pragma unroll
      for (int e = 0; e < 4; ++e) {
        int d = ed4 + e;
        float rg = 1.f / (1.f + __expf(-(ir4[e] + Gh[d][eb])));
        float u  = 1.f / (1.f + __expf(-(iz4[e] + Gh[64 + d][eb])));
        float nn = tanhf(in4[e] + rg * Gh[128 + d][eb]);
        hv4[e] = (1.f - u) * nn + u * hp4[e];
      }
      *(v4f*)&Hf[eb][ed4] = hv4;
      *(v4f*)&out[((((size_t)eb * ROI + roi) * Tt + t) * Hh + hh) * DHh + ed4] = hv4;
    }
#if !HAVE_ASYNC_LDS
    if (t + 1 < Tt && tid == 0)
      __builtin_prefetch(GI + ((size_t)r * Tt + t + 1) * Bsz * G3, 0, 0);
#endif
    __syncthreads();
  }
}

extern "C" void kernel_launch(void* const* d_in, const int* in_sizes, int n_in,
                              void* d_out, int out_size, void* d_ws, size_t ws_size,
                              hipStream_t stream) {
  (void)in_sizes; (void)n_in; (void)out_size; (void)ws_size;
  const float* x   = (const float*)d_in[0];
  const float* a   = (const float*)d_in[1];
  const float* eps = (const float*)d_in[2];
  const float* W1  = (const float*)d_in[3];
  const float* b1  = (const float*)d_in[4];
  const float* g1  = (const float*)d_in[5];
  const float* be1 = (const float*)d_in[6];
  const float* W2  = (const float*)d_in[7];
  const float* b2  = (const float*)d_in[8];
  const float* g2  = (const float*)d_in[9];
  const float* be2 = (const float*)d_in[10];
  const float* Wih = (const float*)d_in[11];
  const float* Whh = (const float*)d_in[12];
  const float* bih = (const float*)d_in[13];
  const float* bhh = (const float*)d_in[14];
  float* out = (float*)d_out;

  const size_t NZ = (size_t)Bsz * ROI * Tt * Hh * Ff;       // 27,648,000
  float* ws = (float*)d_ws;
  float* z1 = ws;
  float* z2 = z1 + NZ;
  float* GI = z2 + NZ;                                       // R*T*B*192
  float* sums = GI + (size_t)Rr * Tt * Bsz * G3;
  float *sum1 = sums,        *sq1 = sums + 192;
  float *sum2 = sums + 384,  *sq2 = sums + 576;
  float *A1 = sums + 768,  *B1c = sums + 960;
  float *A2 = sums + 1152, *B2c = sums + 1344;

  const float invN = 1.f / (float)(Bsz * ROI * Tt);

  k_zero<<<dim3(3), 256, 0, stream>>>(sums, 768);
  k1_gnn1<<<dim3(Bsz * Tt * Hh), 256, 0, stream>>>(x, a, eps, W1, b1, z1, sum1, sq1);
  k_bnfin<<<1, 192, 0, stream>>>(sum1, sq1, g1, be1, A1, B1c, invN);
  k3_gnn2<<<dim3(Bsz * Tt * Hh), 256, 0, stream>>>(z1, A1, B1c, W2, b2, z2, sum2, sq2);
  k_bnfin<<<1, 192, 0, stream>>>(sum2, sq2, g2, be2, A2, B2c, invN);
  k5a_gi<<<dim3(Rr, Tt / 8), 256, 0, stream>>>(z2, A2, B2c, Wih, bih, GI);
  k5b_gru<<<dim3(Rr), 128, 0, stream>>>(GI, Whh, bhh, out);
}
